// EncDecBiLSTMSkipNet6_35923106464134
// MI455X (gfx1250) — compile-verified, matrix operand order checked by reading, not run on hardware
//
#include <hip/hip_runtime.h>
#include <hip/hip_fp16.h>
#include <math.h>

typedef __attribute__((ext_vector_type(16))) _Float16 v16h;
typedef __attribute__((ext_vector_type(8)))  float    v8f;
typedef __attribute__((ext_vector_type(4)))  float    v4f;

#define BATCH   256
#define TFULL   8192
#define T1 8190   // conv1 out length
#define T2 8188   // conv2 out length
#define T3 8190   // deconv2 out length
#define T4 8192   // deconv1 out length

// Workspace layout (floats), channel-last with padded rows:
//  buf1: x1           (B, 8200 rows, 4 ch)
//  cat1: [x2 | rx2]   (B, 8200 rows, 8 ch), rows 0,1 and >=8190 zero
//  cat2: [x3 | x1r]   (B, 8200 rows, 8 ch), rows 0,1 and >=8192 zero
//  afrag: 4 layers x 32 lanes x 16 f16 (precomputed WMMA A fragments)
#define ROWS    8200
#define S1      (ROWS * 4)   // 32800
#define SC      (ROWS * 8)   // 65600
static constexpr size_t OFF_BUF1  = 0;
static constexpr size_t OFF_CAT1  = (size_t)BATCH * S1;
static constexpr size_t OFF_CAT2  = OFF_CAT1 + (size_t)BATCH * SC;
static constexpr size_t OFF_AFRAG = OFF_CAT2 + (size_t)BATCH * SC;  // 41,984,000

#define NTILES          512
#define TILES_PER_WAVE  8
#define GEMM_BLOCKS     ((BATCH * NTILES) / (TILES_PER_WAVE * 8))   // 2048

// ---------------------------------------------------------------------------
// Zero halo/pad rows so the K=32 WMMA dot product never sees non-finite data.
// ---------------------------------------------------------------------------
__global__ void zero_pads(float* __restrict__ ws) {
  const int b = blockIdx.x;
  const int t = threadIdx.x;
  float* b1 = ws + OFF_BUF1 + (size_t)b * S1;
  float* c1 = ws + OFF_CAT1 + (size_t)b * SC;
  float* c2 = ws + OFF_CAT2 + (size_t)b * SC;
  if (t < 16) { c1[t] = 0.f; c2[t] = 0.f; }                       // front 2 rows
  for (int i = t; i < (ROWS - T1) * 4;       i += 256) b1[T1 * 4 + i]       = 0.f;
  for (int i = t; i < (ROWS - (2 + T2)) * 8; i += 256) c1[(2 + T2) * 8 + i] = 0.f;
  for (int i = t; i < (ROWS - (2 + T3)) * 8; i += 256) c2[(2 + T3) * 8 + i] = 0.f;
}

// ---------------------------------------------------------------------------
// Build per-lane WMMA A fragments (16-bit 16x32 A-matrix layout) for all four
// conv layers. Layer 0: conv1 (O=4,CI=8,K=3). Layer 1: conv2 (O=4,CI=4,K=3).
// Layers 2,3: transposed convs, weight (I=8,O=4,K=3) -> wf[o][i][k]=w[i][o][2-k].
// A index: kp = k*CI + i, zero-padded for m>=4 and kp>=KMAX.
// ---------------------------------------------------------------------------
__global__ void build_afrags(float* __restrict__ ws,
                             const float* __restrict__ c1w, const float* __restrict__ c2w,
                             const float* __restrict__ d2w, const float* __restrict__ d1w) {
  const int tid   = threadIdx.x;        // 0..127
  const int layer = tid >> 5;
  const int lane  = tid & 31;
  const int m     = lane & 15;
  const int kb    = (lane < 16) ? 0 : 8;
  const float* wgt = (layer == 0) ? c1w : (layer == 1) ? c2w : (layer == 2) ? d2w : d1w;
  const int  CI     = (layer == 1) ? 4 : 8;
  const int  KMAX   = (layer == 1) ? 12 : 24;
  const bool deconv = layer >= 2;
  v16h a;
#pragma unroll
  for (int j = 0; j < 16; ++j) {
    const int kp = ((j >> 3) << 4) + kb + (j & 7);   // K index 0..31
    float av = 0.f;
    if (m < 4 && kp < KMAX) {
      if (!deconv) av = wgt[(m * CI + kp % CI) * 3 + kp / CI];
      else         av = wgt[((kp & 7) * 4 + m) * 3 + (2 - (kp >> 3))];
    }
    a[j] = (_Float16)av;
  }
  ((v16h*)(ws + OFF_AFRAG))[layer * 32 + lane] = a;
}

// ---------------------------------------------------------------------------
// Conv-as-GEMM via v_wmma_f32_16x16x32_f16.
// B column t = CI*3 contiguous floats at src[b] + t*CI; per lane exactly two
// 8-float groups -> four global_load_b128, no predication. K'>=24 is zero in A
// for every layer, so clamped/garbage (finite) values there are don't-cares.
// ---------------------------------------------------------------------------
template<int LAYER, int CI, bool GUARD, bool RELU, int DSTMODE, int PITCH, int DOFF>
__global__ __launch_bounds__(256) void conv_wmma(
    const float* __restrict__ src, size_t srcStride,
    const float* __restrict__ afragBase, const float* __restrict__ bias,
    float* __restrict__ dst, size_t dstStride, int Tout)
{
  const int lane = threadIdx.x & 31;
  const int wave = (int)blockIdx.x * 8 + (threadIdx.x >> 5);
  const int m    = lane & 15;
  const int kb   = (lane < 16) ? 0 : 8;

  const v16h a = ((const v16h*)afragBase)[LAYER * 32 + lane];
  const float bv0 = bias[0], bv1 = bias[1], bv2 = bias[2], bv3 = bias[3];

  for (int it = 0; it < TILES_PER_WAVE; ++it) {
    const int g = wave * TILES_PER_WAVE + it;
    const int b = g >> 9;                   // / NTILES
    const int t = ((g & 511) << 4) + m;     // this lane's output column
    const float* col = src + (size_t)b * srcStride;

    // Clamp only for the raw-x layer (no pad rows there); ws layers are padded.
    const int tc = GUARD ? (t < Tout - 1 ? t : Tout - 1) : t;
    size_t o0 = (size_t)tc * CI + kb;        // K' = kb..kb+7  (always < 24)
    size_t o1 = (size_t)tc * CI + 16 + kb;   // K' = 16+kb .. 23+kb
    if (GUARD && o1 > (size_t)(TFULL * 8 - 8)) o1 = (size_t)(TFULL * 8 - 8);

    const v4f q0 = *(const v4f*)(col + o0);
    const v4f q1 = *(const v4f*)(col + o0 + 4);
    const v4f q2 = *(const v4f*)(col + o1);
    const v4f q3 = *(const v4f*)(col + o1 + 4);

    v16h bf;
#pragma unroll
    for (int e = 0; e < 4; ++e) {
      bf[e]      = (_Float16)q0[e];
      bf[4 + e]  = (_Float16)q1[e];
      bf[8 + e]  = (_Float16)q2[e];
      bf[12 + e] = (_Float16)q3[e];
    }

    v8f c = {};
    c = __builtin_amdgcn_wmma_f32_16x16x32_f16(false, a, false, bf, (short)0, c,
                                               false, false);

    if (lane < 16 && t < Tout) {
      float r0 = c[0] + bv0, r1 = c[1] + bv1, r2 = c[2] + bv2, r3 = c[3] + bv3;
      if (RELU) {
        r0 = fmaxf(r0, 0.f); r1 = fmaxf(r1, 0.f);
        r2 = fmaxf(r2, 0.f); r3 = fmaxf(r3, 0.f);
      }
      if (DSTMODE == 0) {
        v4f rv = {r0, r1, r2, r3};
        *(v4f*)(dst + (size_t)b * dstStride + (size_t)t * PITCH + DOFF) = rv;
      } else {
        float* p = dst + (size_t)b * dstStride + t;
        p[0] = r0; p[TFULL] = r1; p[2 * TFULL] = r2; p[3 * TFULL] = r3;
      }
    }
  }
}

// ---------------------------------------------------------------------------
// Both IndRNN layers: 2048 independent chains, one lane each, next-x load
// software-pipelined behind the tanh dependency chain.
// ---------------------------------------------------------------------------
__global__ __launch_bounds__(256) void indrnn_kernel(
    float* __restrict__ ws,
    const float* __restrict__ w1, const float* __restrict__ u1, const float* __restrict__ bb1,
    const float* __restrict__ w2, const float* __restrict__ u2, const float* __restrict__ bb2)
{
  const int tid   = blockIdx.x * blockDim.x + threadIdx.x;
  const int c     = tid & 3;
  const int b     = (tid >> 2) & (BATCH - 1);
  const int which = tid >> 10;   // uniform per wave

  const float* in; float* out; int is, T; float w, u, bb;
  if (which == 0) {
    in  = ws + OFF_BUF1 + (size_t)b * S1 + c;            is = 4; T = T1;
    out = ws + OFF_CAT2 + (size_t)b * SC + 16 + 4 + c;   // rows 2.., ch 4+c
    w = w1[c]; u = u1[c]; bb = bb1[c];
  } else {
    in  = ws + OFF_CAT1 + (size_t)b * SC + 16 + c;       is = 8; T = T2;
    out = ws + OFF_CAT1 + (size_t)b * SC + 16 + 4 + c;
    w = w2[c]; u = u2[c]; bb = bb2[c];
  }

  float h  = 0.f;
  float xv = in[0];
  for (int t = 0; t < T; ++t) {
    const float xn = (t + 1 < T) ? in[(size_t)(t + 1) * is] : 0.f;
    h = tanhf(fmaf(u, h, fmaf(w, xv, bb)));
    out[(size_t)t * 8] = h;
    xv = xn;
  }
}

// ---------------------------------------------------------------------------
extern "C" void kernel_launch(void* const* d_in, const int* in_sizes, int n_in,
                              void* d_out, int out_size, void* d_ws, size_t ws_size,
                              hipStream_t stream) {
  (void)in_sizes; (void)n_in; (void)out_size; (void)ws_size;
  const float* x   = (const float*)d_in[0];
  const float* c1w = (const float*)d_in[1];
  const float* c1b = (const float*)d_in[2];
  const float* c2w = (const float*)d_in[3];
  const float* c2b = (const float*)d_in[4];
  const float* r1w = (const float*)d_in[5];
  const float* r1u = (const float*)d_in[6];
  const float* r1b = (const float*)d_in[7];
  const float* r2w = (const float*)d_in[8];
  const float* r2u = (const float*)d_in[9];
  const float* r2b = (const float*)d_in[10];
  const float* d2w = (const float*)d_in[11];
  const float* d2b = (const float*)d_in[12];
  const float* d1w = (const float*)d_in[13];
  const float* d1b = (const float*)d_in[14];
  float* out = (float*)d_out;
  float* ws  = (float*)d_ws;
  const float* af = ws + OFF_AFRAG;

  zero_pads<<<BATCH, 256, 0, stream>>>(ws);
  build_afrags<<<1, 128, 0, stream>>>(ws, c1w, c2w, d2w, d1w);

  // conv1: x (B,T,8) -> x1 (buf1, channel-last), relu
  conv_wmma<0, 8, true,  true,  0, 4, 0 ><<<GEMM_BLOCKS, 256, 0, stream>>>(
      x, (size_t)TFULL * 8, af, c1b, ws + OFF_BUF1, S1, T1);

  // conv2: x1 -> x2 into cat1 ch 0..3 (rows offset by 2 halo rows), relu
  conv_wmma<1, 4, false, true,  0, 8, 16><<<GEMM_BLOCKS, 256, 0, stream>>>(
      ws + OFF_BUF1, S1, af, c2b, ws + OFF_CAT1, SC, T2);

  // both IndRNNs: x1->x1r (cat2 ch4..7), x2->rx2 (cat1 ch4..7)
  indrnn_kernel<<<8, 256, 0, stream>>>(ws, r1w, r1u, r1b, r2w, r2u, r2b);

  // deconv2: cat1=[x2|rx2] -> x3 into cat2 ch 0..3, relu
  conv_wmma<2, 8, false, true,  0, 8, 16><<<GEMM_BLOCKS, 256, 0, stream>>>(
      ws + OFF_CAT1, SC, af, d2b, ws + OFF_CAT2, SC, T3);

  // deconv1: cat2=[x3|x1r] -> x4 straight to d_out in (B,C,T) flat order
  conv_wmma<3, 8, false, false, 1, 0, 0 ><<<GEMM_BLOCKS, 256, 0, stream>>>(
      ws + OFF_CAT2, SC, af, d1b, out, (size_t)4 * TFULL, T4);
}
